// GraphConvHyp_79121887527631
// MI455X (gfx1250) — matrix operand order repaired and checked by reading, not compiled, not run on hardware
//
#include <hip/hip_runtime.h>
#include <hip/hip_bf16.h>

typedef __attribute__((ext_vector_type(16))) _Float16 v16h;
typedef __attribute__((ext_vector_type(8)))  _Float16 v8h;
typedef __attribute__((ext_vector_type(4)))  _Float16 v4h;
typedef __attribute__((ext_vector_type(8)))  float    v8f;
typedef __attribute__((ext_vector_type(4)))  float    v4f;

#define NN   8192
#define DIN  128
#define DOUT 128
#define EPSV 1e-7f
#define GMAX 1e7f

// ---------------------------------------------------------------------------
// Kernel 1: row sums of adj -> D[i] = rsqrt(1 + sum_j adj[i,j])
// One block per row, 256 threads stride the 8192 columns with b128 loads.
// ---------------------------------------------------------------------------
__global__ __launch_bounds__(256) void k_rowsum_D(const float* __restrict__ adj,
                                                  float* __restrict__ D) {
    const int row = blockIdx.x;
    const float* a = adj + (size_t)row * NN;
    float s = 0.0f;
    for (int c = threadIdx.x * 4; c < NN; c += 256 * 4) {
        v4f v = *(const v4f*)(a + c);
        s += v.x + v.y + v.z + v.w;
    }
    __shared__ float red[256];
    red[threadIdx.x] = s;
    __syncthreads();
    for (int off = 128; off > 0; off >>= 1) {
        if (threadIdx.x < off) red[threadIdx.x] += red[threadIdx.x + off];
        __syncthreads();
    }
    if (threadIdx.x == 0) D[row] = rsqrtf(1.0f + red[0]);
}

// ---------------------------------------------------------------------------
// Kernel 2: per-row gamma; transposed f16 y:  yT[n][j] = (f16)(D[j]*x[j][n]);
// transposed f16 weights: WhT[n][k] = (f16)W[k][n].
// One block per node row j, 128 threads = columns.
// ---------------------------------------------------------------------------
__global__ __launch_bounds__(128) void k_prep(const float* __restrict__ x,
                                              const float* __restrict__ W,
                                              const float* __restrict__ D,
                                              float* __restrict__ gam,
                                              _Float16* __restrict__ yT,
                                              _Float16* __restrict__ WhT) {
    const int j = blockIdx.x;
    const int t = threadIdx.x;
    const float xv = x[(size_t)j * DIN + t];
    __shared__ float red[128];
    red[t] = xv * xv;
    __syncthreads();
    for (int off = 64; off > 0; off >>= 1) {
        if (t < off) red[t] += red[t + off];
        __syncthreads();
    }
    if (t == 0) {
        const float r2 = red[0];
        gam[j] = fminf(2.0f / (1.0f - r2), GMAX);
    }
    yT[(size_t)t * NN + j] = (_Float16)(D[j] * xv);
    if (j < DIN) {  // WhT[n=j][k=t] = W[k=t][n=j]
        WhT[(size_t)j * DIN + t] = (_Float16)W[(size_t)t * DOUT + j];
    }
}

// ---------------------------------------------------------------------------
// Kernel 3: T = adj @ (D (.) x)  via f16 WMMA, f32 accumulate.
// Also Sd[i] = sum_j adj[i,j]*D[j] fused into the tile load.
// Block: 256 threads (8 waves), output tile M=32 x N=128 -> 2x8 WMMA tiles,
// wave w owns (m_t = w&1, n tiles 2*(w>>1), 2*(w>>1)+1).
// adj tile (32x32) staged f32->f16 in double-buffered LDS.
// B fragments come straight from global yT (contiguous, L2-resident 2 MB).
// ---------------------------------------------------------------------------
__global__ __launch_bounds__(256) void k_spmm(const float* __restrict__ adj,
                                              const _Float16* __restrict__ yT,
                                              const float* __restrict__ D,
                                              float* __restrict__ T,
                                              float* __restrict__ Sd) {
    __shared__ _Float16 tA[2][32 * 32];
    __shared__ float sdred[256];

    const int i0   = blockIdx.x * 32;
    const int t    = threadIdx.x;
    const int w    = t >> 5;
    const int lane = t & 31;
    const int m_t  = w & 1;
    const int n_p  = w >> 1;
    const int lrow = lane & 15;
    const int lhi  = lane >> 4;

    // cooperative tile-load mapping: thread t -> tile row tr, 4 cols at tc
    const int tr = t >> 3;
    const int tc = (t & 7) * 4;
    const float* arow = adj + (size_t)(i0 + tr) * NN + tc;

    // A fragment LDS offset (halves): row m_t*16+lrow, K base 16*lhi (+8 second chunk)
    const int aoff = (m_t * 16 + lrow) * 32 + 16 * lhi;

    // B fragment pointers: column n = 32*n_p + lrow (and +16), K offset 16*lhi
    const _Float16* bp0 = yT + (size_t)(n_p * 32 + lrow) * NN + 16 * lhi;
    const _Float16* bp1 = bp0 + (size_t)16 * NN;

    float sdacc = 0.0f;
    v8f acc0 = {};
    v8f acc1 = {};

    for (int j0 = 0; j0 < NN; j0 += 32) {
        const int buf = (j0 >> 5) & 1;
        // load + convert adj sub-tile, fuse Sd accumulation
        v4f av = *(const v4f*)(arow + j0);
        __builtin_prefetch(arow + j0 + 64, 0, 1);      // global_prefetch_b8
        v4f dv = *(const v4f*)(D + j0 + tc);
        sdacc += av.x * dv.x + av.y * dv.y + av.z * dv.z + av.w * dv.w;
        v4h h;
        h.x = (_Float16)av.x; h.y = (_Float16)av.y;
        h.z = (_Float16)av.z; h.w = (_Float16)av.w;
        *(v4h*)&tA[buf][tr * 32 + tc] = h;
        __syncthreads();

        // A fragment (16x32 f16): halves 0..7 = K base..base+7, 8..15 = K base+8..+15
        v8h lo = *(const v8h*)&tA[buf][aoff];
        v8h hi = *(const v8h*)&tA[buf][aoff + 8];
        v16h afr;
#pragma unroll
        for (int q = 0; q < 8; ++q) { afr[q] = lo[q]; afr[q + 8] = hi[q]; }

        // B fragments: lanes 0-15 K=0..15, lanes 16-31 K=16..31 (contiguous in yT)
        v16h b0 = *(const v16h*)(bp0 + j0);
        v16h b1 = *(const v16h*)(bp1 + j0);

        acc0 = __builtin_amdgcn_wmma_f32_16x16x32_f16(false, afr, false, b0,
                                                      (short)0, acc0, false, false);
        acc1 = __builtin_amdgcn_wmma_f32_16x16x32_f16(false, afr, false, b1,
                                                      (short)0, acc1, false, false);
    }

    // store T: VGPR r -> row M = r + 8*lhi, col N = lrow (+tile offset)
    const int col0 = n_p * 32 + lrow;
#pragma unroll
    for (int r = 0; r < 8; ++r) {
        const int row = i0 + m_t * 16 + 8 * lhi + r;
        T[(size_t)row * DIN + col0]      = acc0[r];
        T[(size_t)row * DIN + col0 + 16] = acc1[r];
    }

    // reduce Sd: 8 threads per tile row
    sdred[t] = sdacc;
    __syncthreads();
    if (t < 32) {
        float s = 0.0f;
#pragma unroll
        for (int q = 0; q < 8; ++q) s += sdred[t * 8 + q];
        Sd[i0 + t] = s;
    }
}

// ---------------------------------------------------------------------------
// Kernel 4: epilogue, one wave per 16 rows.
// aggr_i = c_i*D_i*(D_i*x_i + T_i);  mobius(0.5)+logmap0 -> u = s1*s2*aggr;
// y = relu(u@W + b) via WMMA (u built in-register in A-fragment layout);
// out = expmap0(y).
// Lane L holds row rr=L&15; half h=L>>4 holds cols {32c+16h+0..15}.
// ---------------------------------------------------------------------------
__global__ __launch_bounds__(32) void k_epilogue(const float* __restrict__ x,
                                                 const float* __restrict__ T,
                                                 const float* __restrict__ D,
                                                 const float* __restrict__ gam,
                                                 const float* __restrict__ Sd,
                                                 const _Float16* __restrict__ WhT,
                                                 const float* __restrict__ bvec,
                                                 float* __restrict__ out) {
    __shared__ float ys[16 * 128];

    const int i0 = blockIdx.x * 16;
    const int L  = threadIdx.x;
    const int rr = L & 15;
    const int h  = L >> 4;
    const int ir = i0 + rr;

    const float Di = D[ir];
    const float g  = gam[ir];
    const float sd = Sd[ir];
    const float Si = Di * (Di + sd);                 // normalized row sum of a
    const float ci = g / ((g - 1.0f) * Si);          // coef row scale
    const float s0 = ci * Di;

    // load this lane's 64 aggr values in A-fragment order
    float ax[64];
    const float* xr = x + (size_t)ir * DIN;
    const float* Tr = T + (size_t)ir * DIN;
#pragma unroll
    for (int gdx = 0; gdx < 8; ++gdx) {
        const int c = gdx >> 1, sub = gdx & 1;
        const int col = 32 * c + 16 * h + 8 * sub;
        const int o   = c * 16 + sub * 8;
        v4f x0 = *(const v4f*)(xr + col);
        v4f x1 = *(const v4f*)(xr + col + 4);
        v4f t0 = *(const v4f*)(Tr + col);
        v4f t1 = *(const v4f*)(Tr + col + 4);
        ax[o + 0] = s0 * (Di * x0.x + t0.x);
        ax[o + 1] = s0 * (Di * x0.y + t0.y);
        ax[o + 2] = s0 * (Di * x0.z + t0.z);
        ax[o + 3] = s0 * (Di * x0.w + t0.w);
        ax[o + 4] = s0 * (Di * x1.x + t1.x);
        ax[o + 5] = s0 * (Di * x1.y + t1.y);
        ax[o + 6] = s0 * (Di * x1.z + t1.z);
        ax[o + 7] = s0 * (Di * x1.w + t1.w);
    }

    // row norm: lanes L and L^16 jointly hold the full row
    float ss = 0.0f;
#pragma unroll
    for (int k = 0; k < 64; ++k) ss += ax[k] * ax[k];
    ss += __shfl_xor(ss, 16, 32);
    const float nrm = sqrtf(ss);
    const float ns  = fminf(fmaxf(nrm, EPSV), 1.0f - EPSV);
    const float s1  = tanhf(0.5f * atanhf(ns)) / ns;      // mobius scalar 0.5
    const float nz  = s1 * nrm;
    const float nzs = fminf(fmaxf(nz, EPSV), 1.0f - EPSV);
    const float s2  = atanhf(nzs) / nzs;                  // logmap0
    const float su  = s1 * s2;

    // build u as f16 A fragments (4 K-chunks of 32)
    v16h uf[4];
#pragma unroll
    for (int c = 0; c < 4; ++c)
#pragma unroll
        for (int q = 0; q < 16; ++q)
            uf[c][q] = (_Float16)(su * ax[c * 16 + q]);

    // y = relu(u @ W + b): 8 N-tiles x 4 K-chunks of WMMA
#pragma unroll
    for (int nt = 0; nt < 8; ++nt) {
        v8f acc = {};
#pragma unroll
        for (int c = 0; c < 4; ++c) {
            v16h bfr = *(const v16h*)(WhT + (size_t)(nt * 16 + rr) * DIN + 32 * c + 16 * h);
            acc = __builtin_amdgcn_wmma_f32_16x16x32_f16(false, uf[c], false, bfr,
                                                         (short)0, acc, false, false);
        }
        const int col = nt * 16 + rr;
        const float bv = bvec[col];
#pragma unroll
        for (int r = 0; r < 8; ++r) {
            const int m = 8 * h + r;
            ys[m * 128 + col] = fmaxf(acc[r] + bv, 0.0f);
        }
    }
    __syncthreads();

    // expmap0 per row; lane handles cols [h*64, h*64+64) of row rr
    const int cbase = h * 64;
    float ss2 = 0.0f;
#pragma unroll
    for (int q = 0; q < 64; ++q) {
        const float v = ys[rr * 128 + cbase + q];
        ss2 += v * v;
    }
    ss2 += __shfl_xor(ss2, 16, 32);
    const float ny  = sqrtf(ss2);
    const float nys = fmaxf(ny, EPSV);
    const float tt  = tanhf(nys) / nys;
    float* orow = out + (size_t)ir * DOUT + cbase;
#pragma unroll
    for (int q = 0; q < 64; q += 4) {
        v4f v = *(const v4f*)&ys[rr * 128 + cbase + q];
        v4f o; o.x = tt * v.x; o.y = tt * v.y; o.z = tt * v.z; o.w = tt * v.w;
        *(v4f*)(orow + q) = o;
    }
}

// ---------------------------------------------------------------------------
extern "C" void kernel_launch(void* const* d_in, const int* in_sizes, int n_in,
                              void* d_out, int out_size, void* d_ws, size_t ws_size,
                              hipStream_t stream) {
    const float* x   = (const float*)d_in[0];   // [8192,128]
    const float* adj = (const float*)d_in[1];   // [8192,8192]
    const float* W   = (const float*)d_in[2];   // [128,128]
    const float* bv  = (const float*)d_in[3];   // [128]
    float* out = (float*)d_out;                 // [8192,128]

    char* p = (char*)d_ws;
    float* D   = (float*)p;        p += (size_t)NN * sizeof(float);
    float* gam = (float*)p;        p += (size_t)NN * sizeof(float);
    float* Sd  = (float*)p;        p += (size_t)NN * sizeof(float);
    float* T   = (float*)p;        p += (size_t)NN * DIN * sizeof(float);
    _Float16* yT  = (_Float16*)p;  p += (size_t)DIN * NN * sizeof(_Float16);
    _Float16* WhT = (_Float16*)p;  p += (size_t)DIN * DOUT * sizeof(_Float16);

    k_rowsum_D<<<NN, 256, 0, stream>>>(adj, D);
    k_prep<<<NN, 128, 0, stream>>>(x, W, D, gam, yT, WhT);
    k_spmm<<<NN / 32, 256, 0, stream>>>(adj, yT, D, T, Sd);
    k_epilogue<<<NN / 16, 32, 0, stream>>>(x, T, D, gam, Sd, WhT, bv, out);
}
